// MHAtt_41274635714591
// MI455X (gfx1250) — matrix operand order repaired and checked by reading, not compiled, not run on hardware
//
#include <hip/hip_runtime.h>
#include <hip/hip_bf16.h>

// ---------------------------------------------------------------------------
// Gated MHA for MI455X (gfx1250). All matrix math via v_wmma_f32_16x16x32_bf16
// (wave32 WMMA), f32 accumulate. Tiles staged global->LDS with
// GLOBAL_LOAD_ASYNC_TO_LDS_B128 (ASYNCcnt), double-buffered so copies overlap
// the WMMA chain; B-fragments software-pipelined so LDS latency hides under
// WMMA instead of s_wait_dscnt 0 drains.
// ---------------------------------------------------------------------------

#define ASYNC_STAGE 1
#define DEVFN __device__ __forceinline__

constexpr int Bn  = 8;
constexpr int Sn  = 1024;
constexpr int Dn  = 1024;
constexpr int Hn  = 8;
constexpr int DBn = 128;
constexpr int Mrows = Bn * Sn;        // 8192

typedef __attribute__((ext_vector_type(16))) __bf16 v16bf;
typedef __attribute__((ext_vector_type(8)))  float  v8f;

union ABf  { v16bf v; unsigned u[8]; uint4 q[2]; };
union ACCf { v8f   v; float    f[8]; };

// Workspace layout (units: u16 elements). Big weights stored TRANSPOSED [n][k].
constexpr size_t OFF_WQ  = 0;
constexpr size_t OFF_WK  = OFF_WQ  + (size_t)Dn * Dn;
constexpr size_t OFF_WV  = OFF_WK  + (size_t)Dn * Dn;
constexpr size_t OFF_WM  = OFF_WV  + (size_t)Dn * Dn;
constexpr size_t OFF_WGX = OFF_WM  + (size_t)Dn * Dn;       // [dout][din]
constexpr size_t OFF_WGY = OFF_WGX + (size_t)DBn * DBn;     // [dout][din]
constexpr size_t OFF_QH  = OFF_WGY + (size_t)DBn * DBn;     // [B,H,S,DB]
constexpr size_t OFF_KH  = OFF_QH  + (size_t)Mrows * Dn;    // [B,H,S,DB]
constexpr size_t OFF_VT  = OFF_KH  + (size_t)Mrows * Dn;    // [B,H,DB,S] (transposed V)
constexpr size_t OFF_ATT = OFF_VT  + (size_t)Mrows * Dn;    // [B,S,D]

DEVFN unsigned short f2bf(float f) {
    unsigned u = __builtin_bit_cast(unsigned, f);
    return (unsigned short)((u + 0x7FFFu + ((u >> 16) & 1u)) >> 16);
}
DEVFN float bf2f(unsigned short h) {
    unsigned u = ((unsigned)h) << 16;
    return __builtin_bit_cast(float, u);
}
// A-layout 8-element group order: storage order [k0-7, k8-15, k16-23, k24-31]
DEVFN int grp_swap(int g) { return g; }

DEVFN v8f wmma_bf16(const ABf& a, const ABf& b, const v8f& c) {
    return __builtin_amdgcn_wmma_f32_16x16x32_bf16(
        false, a.v, false, b.v, (short)0, c, false, false);
}

// ---- async global->LDS 16-byte copy (gfx1250) ------------------------------
DEVFN void stage16(unsigned short* lds_dst, const unsigned short* gsrc) {
#if ASYNC_STAGE
    unsigned loff = (unsigned)(uintptr_t)lds_dst;            // addr[31:0] == LDS offset
    unsigned long long ga = (unsigned long long)(uintptr_t)gsrc;
    asm volatile("global_load_async_to_lds_b128 %0, %1, off"
                 :: "v"(loff), "v"(ga) : "memory");
#else
    *reinterpret_cast<uint4*>(lds_dst) = *reinterpret_cast<const uint4*>(gsrc);
#endif
}
DEVFN void stage_fence() {
#if ASYNC_STAGE
#if __has_builtin(__builtin_amdgcn_s_wait_asynccnt)
    __builtin_amdgcn_s_wait_asynccnt(0);
#else
    asm volatile("s_wait_asynccnt 0x0" ::: "memory");
#endif
#endif
}

// A-fragment from a global row (normal [.,k] layout): 4x b128 + register select.
DEVFN ABf afrag_global(const unsigned short* rowp, int half) {
    const uint4* p4 = reinterpret_cast<const uint4*>(rowp);
    uint4 c0 = p4[0], c1 = p4[1], c2 = p4[2], c3 = p4[3];   // k0-7,8-15,16-23,24-31
    ABf a;
    a.q[0] = half ? c2 : c0;   // lanes16-31 need {16-23,24-31}
    a.q[1] = half ? c3 : c1;
    return a;
}

// ---------------------------------------------------------------------------
// Transposing f32 -> bf16 weight conversion: dst[n*dim + k] = bf16(src[k*dim + n])
// ---------------------------------------------------------------------------
__global__ void cvt_transpose_bf16(const float* __restrict__ src,
                                   unsigned short* __restrict__ dst, int dim) {
    __shared__ unsigned short t[64 * 72];
    const int bx = blockIdx.x * 64;   // n base
    const int by = blockIdx.y * 64;   // k base
    for (int i = threadIdx.x; i < 64 * 64; i += 256) {
        int r = i >> 6, c = i & 63;   // r = k-local, c = n-local
        t[c * 72 + r] = f2bf(src[(size_t)(by + r) * dim + bx + c]);
    }
    __syncthreads();
    for (int i = threadIdx.x; i < 64 * 64; i += 256) {
        int r = i >> 6, c = i & 63;   // r = n-local, c = k-local
        dst[(size_t)(bx + r) * dim + by + c] = t[r * 72 + c];
    }
}

// ---------------------------------------------------------------------------
// Projection GEMM: P = X(f32)[M,1024] @ W + bias, W given transposed bf16 [n][k].
// VT=false: out bf16 [B,H,S,DB];  VT=true: out bf16 [B,H,DB,S] (for V).
// 128x128 tile, 8 waves, K-step 32, double-buffered LDS, pipelined B-frags.
// ---------------------------------------------------------------------------
constexpr int TS = 48;   // tile row stride (u16), multiple of 16 -> 32B aligned rows

template <bool VT>
__global__ __launch_bounds__(256) void proj_gemm(
    const float* __restrict__ X, const unsigned short* __restrict__ WT,
    const float* __restrict__ bias, unsigned short* __restrict__ outp) {
    const int tid  = threadIdx.x;
    const int wave = tid >> 5, lane = tid & 31;
    const int half = lane >> 4, l16 = lane & 15;
    const int h  = blockIdx.x;              // N-tile == head
    const int n0 = h * 128;
    const int m0 = blockIdx.y * 128;

    __shared__ __align__(32) unsigned short As[2][128 * TS]; // fragment-ordered A
    __shared__ __align__(32) unsigned short Bs[2][128 * TS]; // [n][k] contiguous

    auto stageA = [&](int kk, int buf) {     // f32 -> bf16, k groups in natural order
        for (int i = tid; i < 128 * 32; i += 256) {
            int r = i >> 5, c = i & 31;
            As[buf][r * TS + (grp_swap(c >> 3) << 3) + (c & 7)] =
                f2bf(X[(size_t)(m0 + r) * Dn + kk + c]);
            if (kk + 32 < Dn)
                __builtin_prefetch(&X[(size_t)(m0 + r) * Dn + kk + 32 + c], 0, 3);
        }
    };
    auto stageB = [&](int kk, int buf) {     // async 16B chunks, 2 per row
        for (int i = tid; i < 128 * 2; i += 256) {
            int n = i >> 1, ch = i & 1;
            stage16(&Bs[buf][n * TS + ch * 16],
                    &WT[(size_t)(n0 + n) * Dn + kk + ch * 16]);
        }
    };

    ACCf acc[8];
#pragma unroll
    for (int nf = 0; nf < 8; ++nf)
#pragma unroll
        for (int j = 0; j < 8; ++j) acc[nf].f[j] = 0.0f;

    stageA(0, 0); stageB(0, 0);
    stage_fence();
    __syncthreads();

    for (int k0 = 0; k0 < Dn; k0 += 32) {
        const int cur = (k0 >> 5) & 1, nxt = cur ^ 1;
        if (k0 + 32 < Dn) { stageA(k0 + 32, nxt); stageB(k0 + 32, nxt); }

        ABf a, bb[2];
        a.v = *reinterpret_cast<const v16bf*>(&As[cur][(wave * 16 + l16) * TS + half * 16]);
        bb[0].v = *reinterpret_cast<const v16bf*>(&Bs[cur][l16 * TS + half * 16]);
#pragma unroll
        for (int nf = 0; nf < 8; ++nf) {
            if (nf < 7)
                bb[(nf + 1) & 1].v = *reinterpret_cast<const v16bf*>(
                    &Bs[cur][((nf + 1) * 16 + l16) * TS + half * 16]);
            acc[nf].v = wmma_bf16(a, bb[nf & 1], acc[nf].v);
        }
        stage_fence();
        __syncthreads();
    }
    // epilogue: bias + bf16 store
#pragma unroll
    for (int nf = 0; nf < 8; ++nf) {
        int db = nf * 16 + l16;
        float bv = bias[n0 + db];
#pragma unroll
        for (int j = 0; j < 8; ++j) {
            int m   = wave * 16 + half * 8 + j;
            int row = m0 + m;
            int b = row >> 10, s = row & (Sn - 1);
            unsigned short val = f2bf(acc[nf].f[j] + bv);
            if (VT)
                outp[(((size_t)b * Hn + h) * DBn + db) * Sn + s] = val;
            else
                outp[(((size_t)b * Hn + h) * Sn + s) * DBn + db] = val;
        }
    }
}

// ---------------------------------------------------------------------------
// Output GEMM: out(f32)[M,1024] = A(bf16)[M,1024] @ W (transposed bf16) + bias
// Fully async-staged A and B, double-buffered.
// ---------------------------------------------------------------------------
__global__ __launch_bounds__(256) void out_gemm(
    const unsigned short* __restrict__ Abf, const unsigned short* __restrict__ WT,
    const float* __restrict__ bias, float* __restrict__ out) {
    const int tid  = threadIdx.x;
    const int wave = tid >> 5, lane = tid & 31;
    const int half = lane >> 4, l16 = lane & 15;
    const int n0 = blockIdx.x * 128;
    const int m0 = blockIdx.y * 128;

    __shared__ __align__(32) unsigned short As[2][128 * TS];
    __shared__ __align__(32) unsigned short Bs[2][128 * TS];

    auto stageA = [&](int kk, int buf) {   // async, 8-u16 chunks in natural k order
        for (int i = tid; i < 128 * 4; i += 256) {
            int r = i >> 2, g = i & 3;
            stage16(&As[buf][r * TS + (grp_swap(g) << 3)],
                    &Abf[(size_t)(m0 + r) * Dn + kk + g * 8]);
        }
    };
    auto stageB = [&](int kk, int buf) {
        for (int i = tid; i < 128 * 2; i += 256) {
            int n = i >> 1, ch = i & 1;
            stage16(&Bs[buf][n * TS + ch * 16],
                    &WT[(size_t)(n0 + n) * Dn + kk + ch * 16]);
        }
    };

    ACCf acc[8];
#pragma unroll
    for (int nf = 0; nf < 8; ++nf)
#pragma unroll
        for (int j = 0; j < 8; ++j) acc[nf].f[j] = 0.0f;

    stageA(0, 0); stageB(0, 0);
    stage_fence();
    __syncthreads();

    for (int k0 = 0; k0 < Dn; k0 += 32) {
        const int cur = (k0 >> 5) & 1, nxt = cur ^ 1;
        if (k0 + 32 < Dn) { stageA(k0 + 32, nxt); stageB(k0 + 32, nxt); }

        ABf a, bb[2];
        a.v = *reinterpret_cast<const v16bf*>(&As[cur][(wave * 16 + l16) * TS + half * 16]);
        bb[0].v = *reinterpret_cast<const v16bf*>(&Bs[cur][l16 * TS + half * 16]);
#pragma unroll
        for (int nf = 0; nf < 8; ++nf) {
            if (nf < 7)
                bb[(nf + 1) & 1].v = *reinterpret_cast<const v16bf*>(
                    &Bs[cur][((nf + 1) * 16 + l16) * TS + half * 16]);
            acc[nf].v = wmma_bf16(a, bb[nf & 1], acc[nf].v);
        }
        stage_fence();
        __syncthreads();
    }
#pragma unroll
    for (int nf = 0; nf < 8; ++nf) {
        int n = n0 + nf * 16 + l16;
        float bv = bias[n];
#pragma unroll
        for (int j = 0; j < 8; ++j) {
            int m = wave * 16 + half * 8 + j;
            out[(size_t)(m0 + m) * Dn + n] = acc[nf].f[j] + bv;
        }
    }
}

// ---------------------------------------------------------------------------
// Gate: GX = KH@WgX+bgX, GY = QH@WgY+bgY, g = sigmoid((GX*GY)@Wg2+bg2);
// scale kh *= g0, qh *= g1 in place. Weights tiny (L2-hot) -> global-direct.
// ---------------------------------------------------------------------------
__global__ __launch_bounds__(256) void gate_kernel(
    unsigned short* __restrict__ qh, unsigned short* __restrict__ kh,
    const unsigned short* __restrict__ wgxT, const unsigned short* __restrict__ wgyT,
    const float* __restrict__ bgX, const float* __restrict__ bgY,
    const float* __restrict__ wg2, const float* __restrict__ bg2) {
    const int tid  = threadIdx.x;
    const int wave = tid >> 5, lane = tid & 31;
    const int half = lane >> 4, l16 = lane & 15;
    const int st = blockIdx.x;           // s-tile (8)
    const int bh = blockIdx.y;           // b*H + h (64)

    unsigned short* khp = kh + (size_t)bh * Sn * DBn;
    unsigned short* qhp = qh + (size_t)bh * Sn * DBn;

    ACCf gx[8], gy[8];
#pragma unroll
    for (int nf = 0; nf < 8; ++nf)
#pragma unroll
        for (int j = 0; j < 8; ++j) { gx[nf].f[j] = 0.f; gy[nf].f[j] = 0.f; }

    const int rowA = st * 128 + wave * 16 + l16;
#pragma unroll
    for (int ks = 0; ks < 4; ++ks) {
        const int k0 = ks * 32;
        ABf aK = afrag_global(&khp[(size_t)rowA * DBn + k0], half);
        ABf aQ = afrag_global(&qhp[(size_t)rowA * DBn + k0], half);
#pragma unroll
        for (int nf = 0; nf < 8; ++nf) {
            int n = nf * 16 + l16;
            ABf bX, bY;
            bX.v = *reinterpret_cast<const v16bf*>(&wgxT[(size_t)n * DBn + k0 + half * 16]);
            bY.v = *reinterpret_cast<const v16bf*>(&wgyT[(size_t)n * DBn + k0 + half * 16]);
            gx[nf].v = wmma_bf16(aK, bX, gx[nf].v);
            gy[nf].v = wmma_bf16(aQ, bY, gy[nf].v);
        }
    }

    float t0[8], t1[8];
#pragma unroll
    for (int j = 0; j < 8; ++j) { t0[j] = 0.f; t1[j] = 0.f; }
#pragma unroll
    for (int nf = 0; nf < 8; ++nf) {
        int n = nf * 16 + l16;
        float bx = bgX[n], by = bgY[n];
        float w0 = wg2[n * 2 + 0], w1 = wg2[n * 2 + 1];
#pragma unroll
        for (int j = 0; j < 8; ++j) {
            float p = (gx[nf].f[j] + bx) * (gy[nf].f[j] + by);
            t0[j] += p * w0;
            t1[j] += p * w1;
        }
    }
#pragma unroll
    for (int j = 0; j < 8; ++j)
        for (int msk = 1; msk < 16; msk <<= 1) {
            t0[j] += __shfl_xor(t0[j], msk, 16);
            t1[j] += __shfl_xor(t1[j], msk, 16);
        }

    __shared__ float gbuf[8][16][2];
    if (l16 == 0) {
        float b20 = bg2[0], b21 = bg2[1];
#pragma unroll
        for (int j = 0; j < 8; ++j) {
            gbuf[wave][half * 8 + j][0] = 1.0f / (1.0f + __expf(-(t0[j] + b20)));
            gbuf[wave][half * 8 + j][1] = 1.0f / (1.0f + __expf(-(t1[j] + b21)));
        }
    }
    __syncthreads();

    for (int r = 0; r < 16; ++r) {
        float g0 = gbuf[wave][r][0], g1 = gbuf[wave][r][1];
        size_t base = ((size_t)bh * Sn + st * 128 + wave * 16 + r) * DBn;
        for (int c = lane; c < DBn; c += 32) {
            kh[base + c] = f2bf(bf2f(kh[base + c]) * g0);
            qh[base + c] = f2bf(bf2f(qh[base + c]) * g1);
        }
    }
}

// ---------------------------------------------------------------------------
// Flash attention per (b,h,q-tile of 128). K and V^T staged async together
// into separate LDS buffers (one fence+barrier per kv tile); both copies land
// before the scores WMMAs start. P transposed via per-wave LDS.
// ---------------------------------------------------------------------------
constexpr int KVS = 144;  // padded stride, multiple of 16
__global__ __launch_bounds__(256) void attn_kernel(
    const unsigned short* __restrict__ qh, const unsigned short* __restrict__ kh,
    const unsigned short* __restrict__ vT, const unsigned char* __restrict__ mask,
    unsigned short* __restrict__ attout) {
    extern __shared__ __align__(32) unsigned short smem[];
    unsigned short* k_lds = smem;                    // [128][KVS]
    unsigned short* v_lds = smem + 128 * KVS;        // [128][KVS] (V^T: [d][kv])
    unsigned short* p_lds = smem + 2 * 128 * KVS;    // [8 waves][16][KVS]

    const int tid  = threadIdx.x;
    const int wave = tid >> 5, lane = tid & 31;
    const int half = lane >> 4, l16 = lane & 15;
    const int qt = blockIdx.x;
    const int bh = blockIdx.y;
    const int b = bh >> 3, h = bh & 7;

    const unsigned short* qp  = qh + (size_t)bh * Sn * DBn;   // [S][DB]
    const unsigned short* kp  = kh + (size_t)bh * Sn * DBn;   // [S][DB]
    const unsigned short* vtp = vT + (size_t)bh * DBn * Sn;   // [DB][S]

    // Q fragments straight from global (L2-hot after proj): 4x b128 + select
    ABf qfrag[4];
    {
        int qrow = qt * 128 + wave * 16 + l16;
#pragma unroll
        for (int ks = 0; ks < 4; ++ks)
            qfrag[ks] = afrag_global(&qp[(size_t)qrow * DBn + ks * 32], half);
    }

    ACCf acc[8];
    float mrow[8], lrow[8];
#pragma unroll
    for (int nf = 0; nf < 8; ++nf)
#pragma unroll
        for (int j = 0; j < 8; ++j) acc[nf].f[j] = 0.f;
#pragma unroll
    for (int j = 0; j < 8; ++j) { mrow[j] = -1e30f; lrow[j] = 0.f; }

    const float scale = 0.08838834764831845f;  // 1/sqrt(128)
    unsigned short* pw = p_lds + wave * 16 * KVS;

    for (int kv0 = 0; kv0 < Sn; kv0 += 128) {
        __syncthreads();   // previous iteration's reads of k_lds/v_lds done
        for (int i = tid; i < 128 * 16; i += 256) {   // K [kv][d]
            int r = i >> 4, ch = i & 15;
            stage16(&k_lds[r * KVS + ch * 8],
                    &kp[(size_t)(kv0 + r) * DBn + ch * 8]);
        }
        for (int i = tid; i < 128 * 16; i += 256) {   // V^T [d][kv]
            int d = i >> 4, ch = i & 15;
            stage16(&v_lds[d * KVS + ch * 8],
                    &vtp[(size_t)d * Sn + kv0 + ch * 8]);
        }
        stage_fence();
        __syncthreads();

        ACCf sacc[8];
#pragma unroll
        for (int nf = 0; nf < 8; ++nf)
#pragma unroll
            for (int j = 0; j < 8; ++j) sacc[nf].f[j] = 0.f;

#pragma unroll
        for (int ks = 0; ks < 4; ++ks) {
            ABf bb[2];   // pipelined B-frags: B[k=d][n=kv]
            bb[0].v = *reinterpret_cast<const v16bf*>(
                &k_lds[l16 * KVS + ks * 32 + half * 16]);
#pragma unroll
            for (int nf = 0; nf < 8; ++nf) {
                if (nf < 7)
                    bb[(nf + 1) & 1].v = *reinterpret_cast<const v16bf*>(
                        &k_lds[((nf + 1) * 16 + l16) * KVS + ks * 32 + half * 16]);
                sacc[nf].v = wmma_bf16(qfrag[ks], bb[nf & 1], sacc[nf].v);
            }
        }

        // scale + mask + online softmax
#pragma unroll
        for (int nf = 0; nf < 8; ++nf) {
            int kvc = kv0 + nf * 16 + l16;
            bool mk = mask[(size_t)b * Sn + kvc] != 0;
#pragma unroll
            for (int j = 0; j < 8; ++j) {
                float sv = sacc[nf].f[j] * scale;
                sacc[nf].f[j] = mk ? -1e9f : sv;
            }
        }
        float mnew[8], alpha[8];
#pragma unroll
        for (int j = 0; j < 8; ++j) {
            float mx = -1e30f;
#pragma unroll
            for (int nf = 0; nf < 8; ++nf) mx = fmaxf(mx, sacc[nf].f[j]);
            for (int msk = 1; msk < 16; msk <<= 1)
                mx = fmaxf(mx, __shfl_xor(mx, msk, 16));
            mnew[j]  = fmaxf(mrow[j], mx);
            alpha[j] = __expf(mrow[j] - mnew[j]);
            mrow[j]  = mnew[j];
        }
        float rsum[8];
#pragma unroll
        for (int j = 0; j < 8; ++j) rsum[j] = 0.f;
#pragma unroll
        for (int nf = 0; nf < 8; ++nf)
#pragma unroll
            for (int j = 0; j < 8; ++j) {
                float p = __expf(sacc[nf].f[j] - mnew[j]);
                sacc[nf].f[j] = p;
                rsum[j] += p;
            }
#pragma unroll
        for (int j = 0; j < 8; ++j) {
            for (int msk = 1; msk < 16; msk <<= 1)
                rsum[j] += __shfl_xor(rsum[j], msk, 16);
            lrow[j] = lrow[j] * alpha[j] + rsum[j];
        }
#pragma unroll
        for (int nf = 0; nf < 8; ++nf)
#pragma unroll
            for (int j = 0; j < 8; ++j) acc[nf].f[j] *= alpha[j];

        // stage P (bf16) into per-wave LDS, columns in natural order
#pragma unroll
        for (int nf = 0; nf < 8; ++nf) {
#pragma unroll
            for (int j = 0; j < 8; ++j) {
                int c = nf * 16 + l16;
                int w32 = c & 31;
                int pos = (c & ~31) + (grp_swap(w32 >> 3) << 3) + (w32 & 7);
                pw[(half * 8 + j) * KVS + pos] = f2bf(sacc[nf].f[j]);
            }
        }

        // acc += P @ V   (v_lds already staged & fenced; p_lds is wave-private)
#pragma unroll
        for (int ks = 0; ks < 4; ++ks) {
            ABf pa, bb[2];
            pa.v = *reinterpret_cast<const v16bf*>(
                &pw[l16 * KVS + ks * 32 + half * 16]);
            bb[0].v = *reinterpret_cast<const v16bf*>(
                &v_lds[l16 * KVS + ks * 32 + half * 16]);
#pragma unroll
            for (int nf = 0; nf < 8; ++nf) {
                if (nf < 7)
                    bb[(nf + 1) & 1].v = *reinterpret_cast<const v16bf*>(
                        &v_lds[((nf + 1) * 16 + l16) * KVS + ks * 32 + half * 16]);
                acc[nf].v = wmma_bf16(pa, bb[nf & 1], acc[nf].v);
            }
        }
    }

    // epilogue: normalize, store bf16 to [B,S,D] for the output GEMM
#pragma unroll
    for (int nf = 0; nf < 8; ++nf)
#pragma unroll
        for (int j = 0; j < 8; ++j) {
            int m = half * 8 + j;
            int srow = qt * 128 + wave * 16 + m;
            float val = acc[nf].f[j] / lrow[j];
            attout[((size_t)b * Sn + srow) * Dn + h * DBn + nf * 16 + l16] = f2bf(val);
        }
}

// ---------------------------------------------------------------------------
extern "C" void kernel_launch(void* const* d_in, const int* in_sizes, int n_in,
                              void* d_out, int out_size, void* d_ws, size_t ws_size,
                              hipStream_t stream) {
    (void)in_sizes; (void)n_in; (void)out_size; (void)ws_size;
    const float* v    = (const float*)d_in[0];
    const float* k    = (const float*)d_in[1];
    const float* q    = (const float*)d_in[2];
    const unsigned char* mask = (const unsigned char*)d_in[3];
    const float* Wv   = (const float*)d_in[4];
    const float* bv   = (const float*)d_in[5];
    const float* Wk   = (const float*)d_in[6];
    const float* bk   = (const float*)d_in[7];
    const float* Wq   = (const float*)d_in[8];
    const float* bq   = (const float*)d_in[9];
    const float* Wm   = (const float*)d_in[10];
    const float* bm   = (const float*)d_in[11];
    const float* WgX  = (const float*)d_in[12];
    const float* bgX  = (const float*)d_in[13];
    const float* WgY  = (const float*)d_in[14];
    const float* bgY  = (const float*)d_in[15];
    const float* Wg2  = (const float*)d_in[16];
    const float* bg2  = (const float*)d_in[17];
    unsigned short* ws = (unsigned short*)d_ws;
    float* out = (float*)d_out;

    // 1) weights -> transposed bf16
    dim3 tg(Dn / 64, Dn / 64), blk(256);
    cvt_transpose_bf16<<<tg, blk, 0, stream>>>(Wq, ws + OFF_WQ, Dn);
    cvt_transpose_bf16<<<tg, blk, 0, stream>>>(Wk, ws + OFF_WK, Dn);
    cvt_transpose_bf16<<<tg, blk, 0, stream>>>(Wv, ws + OFF_WV, Dn);
    cvt_transpose_bf16<<<tg, blk, 0, stream>>>(Wm, ws + OFF_WM, Dn);
    dim3 tg2(DBn / 64, DBn / 64);
    cvt_transpose_bf16<<<tg2, blk, 0, stream>>>(WgX, ws + OFF_WGX, DBn);
    cvt_transpose_bf16<<<tg2, blk, 0, stream>>>(WgY, ws + OFF_WGY, DBn);

    dim3 grid(Dn / 128, Mrows / 128);   // (8, 64)

    // 2) projections -> bf16 (V stored transposed per head)
    proj_gemm<false><<<grid, blk, 0, stream>>>(q, ws + OFF_WQ, bq, ws + OFF_QH);
    proj_gemm<false><<<grid, blk, 0, stream>>>(k, ws + OFF_WK, bk, ws + OFF_KH);
    proj_gemm<true ><<<grid, blk, 0, stream>>>(v, ws + OFF_WV, bv, ws + OFF_VT);

    // 3) gating (scales kh/qh in place)
    dim3 ggrid(Sn / 128, Bn * Hn);      // (8, 64)
    gate_kernel<<<ggrid, blk, 0, stream>>>(ws + OFF_QH, ws + OFF_KH,
                                           ws + OFF_WGX, ws + OFF_WGY,
                                           bgX, bgY, Wg2, bg2);

    // 4) flash attention (110 KB dynamic LDS: K + V^T + P buffers)
    size_t smem_bytes = (size_t)(2 * 128 * KVS + 8 * 16 * KVS) * sizeof(unsigned short);
    attn_kernel<<<ggrid, blk, smem_bytes, stream>>>(ws + OFF_QH, ws + OFF_KH,
                                                    ws + OFF_VT, mask, ws + OFF_ATT);

    // 5) output projection -> f32
    out_gemm<<<grid, blk, 0, stream>>>(ws + OFF_ATT, ws + OFF_WM, bm, out);
}
